// EdgeGraphEBM_OneStep_1932735283516
// MI455X (gfx1250) — matrix operand order repaired and checked by reading, not compiled, not run on hardware
//
#include <hip/hip_runtime.h>
#include <hip/hip_bf16.h>

typedef __attribute__((ext_vector_type(16))) _Float16 v16h;
typedef __attribute__((ext_vector_type(8)))  float    v8f;

#define BB   2
#define NO   96
#define TT   50
#define SD   4
#define NEg  9120          // NO*(NO-1)
#define FF   64
#define LL   64
#define KK   8
#define NW   15

// ---- workspace (f16) layout -------------------------------------------------
// [0 .. 81920)        : 4 conv weight slots [i*2+cv][f][kk], kk = dk*64 + c  (each 64*320)
// [81920 .. 114688)   : mlp1_fc1_w as [64][512] f16
// [114688 .. 118784)  : mlp1_fc2_w as [64][64]  f16
#define WS_RB   0
#define WS_WT1  81920
#define WS_WT2  114688
#define WS_TOT  118784

// ---- LDS layout (bytes) -----------------------------------------------------
#define XF_OFF 0          // f32 [64][52]  x (channel-major)
#define XT_OFF 13312      // f16 [72][64]  x (time-major, halo rows, zero-padded)
#define HT_OFF 22528      // f16 [72][64]  conv1 output (time-major)
#define SW_OFF 31744      // f16 [20480]   staged weights
#define XW_OFF 72704      // f16 [16][512] window A matrix
#define H1_OFF 89088      // f16 [16][64]  window hidden
#define FB_OFF 91136      // f32 scratch: lat,ga,ba,gb,bb,hv,tv (7*64) + part(128) + xin(416)
#define SMEM_BYTES 95232

__device__ __forceinline__ float swishf(float a) { return a / (1.0f + __expf(-a)); }

// Async global -> LDS copy (CDNA5 Tensor/Async path, ASYNCcnt-tracked).
// LDS destination is the byte address in lds_off; global source is the 64-bit address g (no SGPR base).
__device__ __forceinline__ void async_ld_b128(unsigned lds_off, const void* g) {
  asm volatile("global_load_async_to_lds_b128 %0, %1, off"
               :: "v"(lds_off), "v"(g) : "memory");
}
__device__ __forceinline__ void wait_async0() {
  asm volatile("s_wait_asynccnt 0" ::: "memory");
}

// A fragment: 16x32 f16, row-major source with leading dim ld (halfs).
// lanes 0-15: M=lane, halves0-15 = K k0..k0+15
// lanes16-31: M=lane-16, halves0-15 = K k0+16..k0+31.
__device__ __forceinline__ v16h frag_a(const _Float16* src, int ld, int m0, int k0, int lane) {
  int m  = m0 + (lane & 15);
  int kb = k0 + ((lane >> 4) << 4);
  const _Float16* p = src + m * ld + kb;
  union { v16h v; uint4 q[2]; } u;
  u.q[0] = *(const uint4*)(p);
  u.q[1] = *(const uint4*)(p + 8);
  return u.v;
}

// B fragment: 32x16 f16 from a row-major "W^T-style" buffer wt[n][kdim], ld halfs.
// lane n&15 = column; halves = 16 contiguous K at kb = k0 + (lane>=16)*16.
__device__ __forceinline__ v16h frag_b_rows(const _Float16* wt, int ld, int n0, int k0, int lane) {
  int n  = n0 + (lane & 15);
  int kb = k0 + ((lane >> 4) << 4);
  const _Float16* p = wt + n * ld + kb;
  union { v16h v; uint4 q[2]; } u;
  u.q[0] = *(const uint4*)(p);
  u.q[1] = *(const uint4*)(p + 8);
  return u.v;
}

// Conv B fragment: B[kk][t] = x[c][t+dk-2], kk = dk*64+c, sourced from time-major
// f16 buffer xt[tt][c] (tt = t+2, rows 0..71, halo rows zero). A 16-length K run
// has fixed dk and contiguous c, so it is two b128 LDS loads.
__device__ __forceinline__ v16h frag_b_conv(const _Float16* xt, int n0, int k0, int lane) {
  int t  = n0 + (lane & 15);
  int kb = k0 + ((lane >> 4) << 4);
  int dk = kb >> 6;
  int c0 = kb & 63;
  const _Float16* p = xt + (t + dk) * 64 + c0;
  union { v16h v; uint4 q[2]; } u;
  u.q[0] = *(const uint4*)(p);
  u.q[1] = *(const uint4*)(p + 8);
  return u.v;
}

#define WMMA_F16(A, Bf, C) \
  __builtin_amdgcn_wmma_f32_16x16x32_f16(false, (A), false, (Bf), (short)0, (C), false, false)

// ---- weight pre-conversion kernel ------------------------------------------
__global__ void convert_weights(const float* __restrict__ c1, const float* __restrict__ c2,
                                const float* __restrict__ fc1, const float* __restrict__ fc2,
                                _Float16* __restrict__ ws) {
  int idx = blockIdx.x * blockDim.x + threadIdx.x;
  if (idx < 81920) {
    int slot = idx / 20480, r = idx % 20480;
    int f = r / 320, kk = r % 320;
    int dk = kk >> 6, c = kk & 63;
    int i = slot >> 1, cv = slot & 1;
    const float* src = cv ? c2 : c1;
    ws[idx] = (_Float16)src[((i * 64 + f) * 64 + c) * 5 + dk];
  } else if (idx < WS_WT2) {
    ws[idx] = (_Float16)fc1[idx - WS_WT1];
  } else if (idx < WS_TOT) {
    ws[idx] = (_Float16)fc2[idx - WS_WT2];
  }
}

// ---- fused per-edge kernel --------------------------------------------------
__global__ __launch_bounds__(128)
void edge_ebm_kernel(const float* __restrict__ states, const float* __restrict__ latent,
                     const float* __restrict__ cnnbf_w, const float* __restrict__ cnnbf_b,
                     const float* __restrict__ rb_c1_b, const float* __restrict__ rb_c2_b,
                     const float* __restrict__ rb_l1_w, const float* __restrict__ rb_l1_b,
                     const float* __restrict__ rb_l2_w, const float* __restrict__ rb_l2_b,
                     const float* __restrict__ fc1_b, const float* __restrict__ fc2_b,
                     const float* __restrict__ ml_f1_w, const float* __restrict__ ml_f1_b,
                     const float* __restrict__ ml_f2_w, const float* __restrict__ ml_f2_b,
                     const float* __restrict__ ml_l1_w, const float* __restrict__ ml_l1_b,
                     const float* __restrict__ ml_l2_w, const float* __restrict__ ml_l2_b,
                     const float* __restrict__ em_w, const float* __restrict__ em_b,
                     const int* __restrict__ recv_idx, const int* __restrict__ send_idx,
                     const _Float16* __restrict__ wsf16, float* __restrict__ out) {
  extern __shared__ char smem[];
  float*    s_xf = (float*)(smem + XF_OFF);        // [64][52]
  _Float16* s_xt = (_Float16*)(smem + XT_OFF);     // [72][64]
  _Float16* s_ht = (_Float16*)(smem + HT_OFF);     // [72][64]
  _Float16* s_w  = (_Float16*)(smem + SW_OFF);     // [20480]
  _Float16* s_xw = (_Float16*)(smem + XW_OFF);     // [16][512]
  _Float16* s_h1 = (_Float16*)(smem + H1_OFF);     // [16][64]
  float* fb   = (float*)(smem + FB_OFF);
  float* s_lat = fb;        float* s_ga = fb + 64;  float* s_ba = fb + 128;
  float* s_gb = fb + 192;   float* s_bb = fb + 256; float* s_hv = fb + 320;
  float* s_tv = fb + 384;   float* s_part = fb + 448; float* s_xin = fb + 576; // [8][52]

  const unsigned lds_w = (unsigned)(uintptr_t)(void*)s_w;   // LDS byte offset of s_w

  const int tid  = threadIdx.x;
  const int wave = tid >> 5;
  const int lane = tid & 31;
  const int lo   = lane & 15;
  const int hi   = lane >> 4;
  const int b = blockIdx.x / NEg;
  const int e = blockIdx.x % NEg;

  __builtin_prefetch(wsf16, 0, 0);

  // ---- step 0: zero time-major buffers, load latent, gather edge input -----
  {
    unsigned* zp = (unsigned*)(smem + XT_OFF);     // s_xt + s_ht contiguous: 18432 B
    for (int u = tid; u < 4608; u += 128) zp[u] = 0u;
    for (int u = tid; u < 416; u += 128) s_xin[u] = 0.0f;
    if (tid < 64) s_lat[tid] = latent[(size_t)blockIdx.x * 64 + tid];
    int rn = recv_idx[e], sn = send_idx[e];
    for (int u = tid; u < 8 * TT; u += 128) {
      int c = u / TT, t = u % TT;
      int node = (c < 4) ? rn : sn;
      s_xin[c * 52 + t + 1] = states[(((size_t)b * NO + node) * TT + t) * SD + (c & 3)];
    }
  }
  __syncthreads();

  // ---- step 1: initial conv (8->64, k=3, pad 1) + swish (VALU, tiny) -------
  for (int u = tid; u < FF * TT; u += 128) {
    int f = u / TT, t = u % TT;
    float a = cnnbf_b[f];
#pragma unroll
    for (int c = 0; c < 8; ++c)
#pragma unroll
      for (int dk = 0; dk < 3; ++dk)
        a += cnnbf_w[(f * 8 + c) * 3 + dk] * s_xin[c * 52 + t + dk];
    float v = swishf(a);
    s_xf[f * 52 + t] = v;
    s_xt[(t + 2) * 64 + f] = (_Float16)v;
  }
  __syncthreads();

  // ---- step 2: two conditional residual blocks (WMMA convs) ----------------
  for (int i = 0; i < 2; ++i) {
    // kick off conv1 weight DMA (global -> LDS, ASYNCcnt) so it overlaps the
    // latent FiLM matvec below
    {
      const char* src = (const char*)(wsf16 + (size_t)(i * 2 + 0) * 20480);
      for (int u = tid; u < 2560; u += 128)
        async_ld_b128(lds_w + u * 16, src + (size_t)u * 16);
    }
    // FiLM params from latent (VALU) — overlapped with async DMA
    for (int u = tid; u < 256; u += 128) {
      int sel = u >> 7, j = u & 127;
      const float* w  = sel ? rb_l2_w : rb_l1_w;
      const float* bv = sel ? rb_l2_b : rb_l1_b;
      float a = bv[i * 128 + j];
      for (int c = 0; c < 64; ++c) a += w[((i * 128 + j) << 6) + c] * s_lat[c];
      float* dst = sel ? (j < 64 ? s_gb : s_bb) : (j < 64 ? s_ga : s_ba);
      dst[j & 63] = a;
    }
    wait_async0();
    __syncthreads();
    // conv1: C(64x64) = W(64x320) * im2col(320x64); wave = M tile
    for (int nt = 0; nt < 4; ++nt) {
      v8f acc = {};
#pragma unroll
      for (int ks = 0; ks < 10; ++ks) {
        v16h A  = frag_a(s_w, 320, wave * 16, ks * 32, lane);
        v16h Bf = frag_b_conv(s_xt, nt * 16, ks * 32, lane);
        acc = WMMA_F16(A, Bf, acc);
      }
      int t = nt * 16 + lo;
      if (t < TT) {
#pragma unroll
        for (int v = 0; v < 8; ++v) {
          int f = wave * 16 + v + (hi << 3);
          float val = acc[v] + rb_c1_b[i * 64 + f];
          val = swishf(s_ga[f] * val + s_ba[f]);
          s_ht[(t + 2) * 64 + f] = (_Float16)val;
        }
      }
    }
    __syncthreads();
    // stage conv2 weights via async DMA
    {
      const char* src = (const char*)(wsf16 + (size_t)(i * 2 + 1) * 20480);
      for (int u = tid; u < 2560; u += 128)
        async_ld_b128(lds_w + u * 16, src + (size_t)u * 16);
      wait_async0();
    }
    __syncthreads();
    // conv2 + FiLM + swish + residual
    for (int nt = 0; nt < 4; ++nt) {
      v8f acc = {};
#pragma unroll
      for (int ks = 0; ks < 10; ++ks) {
        v16h A  = frag_a(s_w, 320, wave * 16, ks * 32, lane);
        v16h Bf = frag_b_conv(s_ht, nt * 16, ks * 32, lane);
        acc = WMMA_F16(A, Bf, acc);
      }
      int t = nt * 16 + lo;
      if (t < TT) {
#pragma unroll
        for (int v = 0; v < 8; ++v) {
          int f = wave * 16 + v + (hi << 3);
          float val = acc[v] + rb_c2_b[i * 64 + f];
          val = swishf(s_gb[f] * val + s_bb[f]);
          float xn = s_xf[f * 52 + t] + val;
          s_xf[f * 52 + t] = xn;
          s_xt[(t + 2) * 64 + f] = (_Float16)xn;
        }
      }
    }
    __syncthreads();
  }

  // ---- step 3: windowed MLP (WMMA) -----------------------------------------
  // Build A = xw (15x512, row 15 zero), kdim = f*8+k, xw[w][f*8+k] = x[f][3w+k]
  for (int u = tid; u < 16 * 512; u += 128) {
    int w = u >> 9, kd = u & 511;
    float v = 0.0f;
    if (w < NW) { int f = kd >> 3, k = kd & 7; v = s_xf[f * 52 + w * 3 + k]; }
    s_xw[u] = (_Float16)v;
  }
  v8f acc1 = {};
  for (int ch = 0; ch < 2; ++ch) {
    __syncthreads();
    // stage half of fc1^T (64 rows x 256 K) into s_w via async DMA
    {
      for (int u = tid; u < 2048; u += 128) {
        int f = u >> 5, q = u & 31;
        const char* src = (const char*)(wsf16 + WS_WT1 + (size_t)f * 512 + ch * 256 + q * 8);
        async_ld_b128(lds_w + u * 16, src);
      }
      wait_async0();
    }
    __syncthreads();
#pragma unroll
    for (int ks = 0; ks < 8; ++ks) {
      v16h A  = frag_a(s_xw, 512, 0, ch * 256 + ks * 32, lane);
      v16h Bf = frag_b_rows(s_w, 256, wave * 16, ks * 32, lane);
      acc1 = WMMA_F16(A, Bf, acc1);
    }
  }
  __syncthreads();
  {
    int col = wave * 16 + lo;
#pragma unroll
    for (int v = 0; v < 8; ++v) {
      int w = v + (hi << 3);
      s_h1[w * 64 + col] = (_Float16)swishf(acc1[v] + fc1_b[col]);
    }
  }
  // stage fc2^T (64x64) via async DMA
  {
    const char* src = (const char*)(wsf16 + WS_WT2);
    for (int u = tid; u < 512; u += 128)
      async_ld_b128(lds_w + u * 16, src + (size_t)u * 16);
    wait_async0();
  }
  __syncthreads();
  {
    v8f acc2 = {};
#pragma unroll
    for (int ks = 0; ks < 2; ++ks) {
      v16h A  = frag_a(s_h1, 64, 0, ks * 32, lane);
      v16h Bf = frag_b_rows(s_w, 64, wave * 16, ks * 32, lane);
      acc2 = WMMA_F16(A, Bf, acc2);
    }
    int col = wave * 16 + lo;
    float p = 0.0f;
    int vmax = hi ? 7 : 8;            // exclude padded window row 15
#pragma unroll
    for (int v = 0; v < 8; ++v) if (v < vmax) p += acc2[v];
    s_part[col * 2 + hi] = p;
  }
  __syncthreads();
  if (tid < 64)
    s_hv[tid] = (s_part[tid * 2] + s_part[tid * 2 + 1]) * (1.0f / 15.0f) + fc2_b[tid];
  __syncthreads();

  // ---- step 4: three conditional MLP blocks (VALU, tiny) -------------------
  for (int i = 0; i < 3; ++i) {
    for (int u = tid; u < 256; u += 128) {
      int sel = u >> 7, j = u & 127;
      const float* w  = sel ? ml_l2_w : ml_l1_w;
      const float* bv = sel ? ml_l2_b : ml_l1_b;
      float a = bv[i * 128 + j];
      for (int c = 0; c < 64; ++c) a += w[((i * 128 + j) << 6) + c] * s_lat[c];
      float* dst = sel ? (j < 64 ? s_gb : s_bb) : (j < 64 ? s_ga : s_ba);
      dst[j & 63] = a;
    }
    __syncthreads();
    if (tid < 64) {
      float a = ml_f1_b[i * 64 + tid];
      for (int c = 0; c < 64; ++c) a += ml_f1_w[((i * 64 + tid) << 6) + c] * s_hv[c];
      s_tv[tid] = swishf(s_ga[tid] * a + s_ba[tid]);
    }
    __syncthreads();
    if (tid < 64) {
      float a = ml_f2_b[i * 64 + tid];
      for (int c = 0; c < 64; ++c) a += ml_f2_w[((i * 64 + tid) << 6) + c] * s_tv[c];
      s_hv[tid] += swishf(s_gb[tid] * a + s_bb[tid]);
    }
    __syncthreads();
  }

  // ---- step 5: energy head -------------------------------------------------
  if (tid == 0) {
    float a = em_b[0];
    for (int c = 0; c < 64; ++c) a += em_w[c] * s_hv[c];
    out[blockIdx.x] = a;
  }
}

extern "C" void kernel_launch(void* const* d_in, const int* in_sizes, int n_in,
                              void* d_out, int out_size, void* d_ws, size_t ws_size,
                              hipStream_t stream) {
  (void)in_sizes; (void)n_in; (void)out_size; (void)ws_size;
  const float* states   = (const float*)d_in[0];
  const float* latent   = (const float*)d_in[1];
  const float* cnnbf_w  = (const float*)d_in[2];
  const float* cnnbf_b  = (const float*)d_in[3];
  const float* rb_c1_w  = (const float*)d_in[4];
  const float* rb_c1_b  = (const float*)d_in[5];
  const float* rb_c2_w  = (const float*)d_in[6];
  const float* rb_c2_b  = (const float*)d_in[7];
  const float* rb_l1_w  = (const float*)d_in[8];
  const float* rb_l1_b  = (const float*)d_in[9];
  const float* rb_l2_w  = (const float*)d_in[10];
  const float* rb_l2_b  = (const float*)d_in[11];
  const float* fc1_w    = (const float*)d_in[12];
  const float* fc1_b    = (const float*)d_in[13];
  const float* fc2_w    = (const float*)d_in[14];
  const float* fc2_b    = (const float*)d_in[15];
  const float* ml_f1_w  = (const float*)d_in[16];
  const float* ml_f1_b  = (const float*)d_in[17];
  const float* ml_f2_w  = (const float*)d_in[18];
  const float* ml_f2_b  = (const float*)d_in[19];
  const float* ml_l1_w  = (const float*)d_in[20];
  const float* ml_l1_b  = (const float*)d_in[21];
  const float* ml_l2_w  = (const float*)d_in[22];
  const float* ml_l2_b  = (const float*)d_in[23];
  const float* em_w     = (const float*)d_in[24];
  const float* em_b     = (const float*)d_in[25];
  const int*   recv_idx = (const int*)d_in[26];
  const int*   send_idx = (const int*)d_in[27];
  _Float16* ws = (_Float16*)d_ws;
  float* out = (float*)d_out;

  convert_weights<<<(WS_TOT + 255) / 256, 256, 0, stream>>>(rb_c1_w, rb_c2_w, fc1_w, fc2_w, ws);

  edge_ebm_kernel<<<dim3(BB * NEg), dim3(128), SMEM_BYTES, stream>>>(
      states, latent, cnnbf_w, cnnbf_b, rb_c1_b, rb_c2_b,
      rb_l1_w, rb_l1_b, rb_l2_w, rb_l2_b, fc1_b, fc2_b,
      ml_f1_w, ml_f1_b, ml_f2_w, ml_f2_b, ml_l1_w, ml_l1_b,
      ml_l2_w, ml_l2_b, em_w, em_b, recv_idx, send_idx, ws, out);
}
